// MixtralSparseMoeBlockLora_8289286881432
// MI455X (gfx1250) — matrix operand order repaired and checked by reading, not compiled, not run on hardware
//
#include <hip/hip_runtime.h>

// ---------------- problem constants ----------------
constexpr int H = 4096;
constexpr int F = 14336;
constexpr int E = 8;
constexpr int R = 32;
constexpr int S = 1024;   // B*S tokens

// ---------------- WMMA types / helpers ----------------
typedef __attribute__((ext_vector_type(16))) __bf16 v16bf;
typedef __attribute__((ext_vector_type(8)))  float  v8f;

__device__ __forceinline__ v8f wmma_bf16(v16bf a, v16bf b, v8f c) {
  // 8 args: (neg_a, A, neg_b, B, c_mod, C, reuse_a, reuse_b)
  return __builtin_amdgcn_wmma_f32_16x16x32_bf16(false, a, false, b, (short)0, c,
                                                 false, false);
}

// A fragment: 16x32 bf16, rows M in lanes 0..15 / 16..31, K-major memory.
// lane<16 holds K {0..7, 16..23}, lane>=16 holds K {8..15, 24..31} (ISA 7.12.2)
__device__ __forceinline__ v16bf load_afrag(const __bf16* __restrict__ A, int lda, int lane) {
  int row  = lane & 15;
  int half = lane >> 4;
  const uint4* p0 = reinterpret_cast<const uint4*>(A + row * lda + half * 8);
  const uint4* p1 = reinterpret_cast<const uint4*>(A + row * lda + 16 + half * 8);
  union { uint4 u[2]; v16bf v; } t;
  t.u[0] = *p0;
  t.u[1] = *p1;
  return t.v;
}

// B fragment: 32x16 bf16 (KxN); lane holds column N=lane&15, K-half = lane>>4,
// lane<16 holds K {0..7, 16..23}, lane>=16 holds K {8..15, 24..31} (matches A).
__device__ __forceinline__ v16bf load_bfrag(const __bf16* __restrict__ B, int ldb, int lane) {
  int col = lane & 15;
  int kh  = lane >> 4;
  const uint4* q0 = reinterpret_cast<const uint4*>(B + col * ldb + kh * 8);
  const uint4* q1 = reinterpret_cast<const uint4*>(B + col * ldb + 16 + kh * 8);
  union { uint4 u[2]; v16bf v; } t;
  t.u[0] = *q0;
  t.u[1] = *q1;
  return t.v;
}

// ---------------- elementwise kernels ----------------
__global__ void cvt_f32_bf16_kernel(const float* __restrict__ in,
                                    __bf16* __restrict__ out, size_t n) {
  size_t i = (size_t)blockIdx.x * blockDim.x + threadIdx.x;
  size_t stride = (size_t)gridDim.x * blockDim.x;
  for (; i < n; i += stride) out[i] = (__bf16)in[i];
}

__global__ void zero_f32_kernel(float* __restrict__ p, size_t n) {
  size_t i = (size_t)blockIdx.x * blockDim.x + threadIdx.x;
  size_t stride = (size_t)gridDim.x * blockDim.x;
  for (; i < n; i += stride) p[i] = 0.f;
}

// V2b[t,r] = cw[t,e] * V2[t,r]  (bf16)
__global__ void v2_scale_cvt_kernel(const float* __restrict__ V2,
                                    const float* __restrict__ cw, int e,
                                    __bf16* __restrict__ out) {
  int i = blockIdx.x * 256 + threadIdx.x;   // n = S*R = 32768
  int t = i >> 5;
  out[i] = (__bf16)(V2[i] * cw[t * E + e]);
}

// ---------------- router: logits, softmax, top-2 combine weights ----------------
__global__ __launch_bounds__(256) void router_kernel(const float* __restrict__ x,
                                                     const float* __restrict__ gate_w,
                                                     float* __restrict__ logits_out,
                                                     float* __restrict__ cw) {
  int t   = blockIdx.x;
  int tid = threadIdx.x;
  const float* xr = x + (size_t)t * H;
  float acc[E];
#pragma unroll
  for (int e = 0; e < E; ++e) acc[e] = 0.f;
  for (int h = tid; h < H; h += 256) {
    float xv = xr[h];
#pragma unroll
    for (int e = 0; e < E; ++e) acc[e] += xv * gate_w[e * H + h];
  }
#pragma unroll
  for (int e = 0; e < E; ++e)
    for (int off = 16; off > 0; off >>= 1) acc[e] += __shfl_xor(acc[e], off, 32);
  __shared__ float sm[8][E];
  int lane = tid & 31, wave = tid >> 5;
  if (lane == 0)
    for (int e = 0; e < E; ++e) sm[wave][e] = acc[e];
  __syncthreads();
  if (tid == 0) {
    float lg[E];
    for (int e = 0; e < E; ++e) {
      float s = 0.f;
      for (int w = 0; w < 8; ++w) s += sm[w][e];
      lg[e] = s;
    }
    float mx = lg[0];
    for (int e = 1; e < E; ++e) mx = fmaxf(mx, lg[e]);
    float p[E], sum = 0.f;
    for (int e = 0; e < E; ++e) { p[e] = __expf(lg[e] - mx); sum += p[e]; }
    for (int e = 0; e < E; ++e) p[e] /= sum;
    int i0 = 0;
    for (int e = 1; e < E; ++e) if (p[e] > p[i0]) i0 = e;
    int i1 = (i0 == 0) ? 1 : 0;
    for (int e = 0; e < E; ++e) if (e != i0 && p[e] > p[i1]) i1 = e;
    float d = p[i0] + p[i1];
    for (int e = 0; e < E; ++e) {
      logits_out[t * E + e] = lg[e];
      cw[t * E + e] = (e == i0) ? p[i0] / d : ((e == i1) ? p[i1] / d : 0.f);
    }
  }
}

// ---------------- big WMMA GEMM: C[M,N] = A[M,K] @ B[N,K]^T (+Cadd) ----------------
// block = 256 threads = 8 waves (2 M-waves x 4 N-waves); block tile 64x256.
// Each wave: 32x64 tile = 2 A-frags x 4 B-frags = 8 WMMAs per 32-deep k-step.
// A tile (64x32) staged through LDS, reused by the 4 N-waves.
__global__ __launch_bounds__(256) void gemm_bf16_big(const __bf16* __restrict__ A, int lda,
                                                     const __bf16* __restrict__ B, int ldb,
                                                     const float* __restrict__ Cadd,
                                                     float* __restrict__ C, int ldc, int K) {
  __shared__ __bf16 As[64 * 32];
  int tid  = threadIdx.x;
  int lane = tid & 31, wave = tid >> 5;
  int mw = wave >> 2, nw = wave & 3;
  int m0 = blockIdx.x * 64;
  int n0 = blockIdx.y * 256;
  int wm = m0 + mw * 32;
  int wn = n0 + nw * 64;
  v8f acc[2][4] = {};
  // cooperative A-stage mapping: 64 rows x 32 cols, 8 bf16 (16B) per thread
  int sr = tid >> 2;
  int sc = (tid & 3) * 8;
  const __bf16* Asrc = A + (size_t)(m0 + sr) * lda + sc;
  for (int k0 = 0; k0 < K; k0 += 32) {
    *reinterpret_cast<uint4*>(&As[sr * 32 + sc]) =
        *reinterpret_cast<const uint4*>(Asrc);
    Asrc += 32;
    __syncthreads();
    v16bf a0 = load_afrag(&As[(mw * 32) * 32], 32, lane);
    v16bf a1 = load_afrag(&As[(mw * 32 + 16) * 32], 32, lane);
    const __bf16* Bk = B + (size_t)wn * ldb + k0;
    if (k0 + 64 < K)  // stream next-next B chunk toward L2/L0
      __builtin_prefetch((const void*)(Bk + (size_t)(lane & 15) * ldb + 64), 0, 0);
#pragma unroll
    for (int j = 0; j < 4; ++j) {
      v16bf bf = load_bfrag(Bk + (size_t)(j * 16) * ldb, ldb, lane);
      acc[0][j] = wmma_bf16(a0, bf, acc[0][j]);
      acc[1][j] = wmma_bf16(a1, bf, acc[1][j]);
    }
    __syncthreads();
  }
  int col = lane & 15;
  int rb  = (lane >> 4) * 8;
#pragma unroll
  for (int ii = 0; ii < 2; ++ii) {
#pragma unroll
    for (int j = 0; j < 4; ++j) {
#pragma unroll
      for (int i = 0; i < 8; ++i) {
        size_t idx = (size_t)(wm + ii * 16 + rb + i) * ldc + wn + j * 16 + col;
        float v = acc[ii][j][i];
        if (Cadd) v += Cadd[idx];
        C[idx] = v;
      }
    }
  }
}

// ---------------- skinny WMMA GEMM: one wave per 16x16 tile, optional batch/acc ----
__global__ __launch_bounds__(32) void gemm_bf16_skinny(const __bf16* __restrict__ A,
                                                       long aStrideE, int lda,
                                                       const __bf16* __restrict__ B,
                                                       long bStrideE, int ldb,
                                                       float* __restrict__ C,
                                                       long cStrideE, int ldc,
                                                       int K, int accumulate) {
  int lane = threadIdx.x & 31;
  long e = blockIdx.z;
  const __bf16* Ae = A + e * aStrideE;
  const __bf16* Be = B + e * bStrideE;
  float*        Ce = C + e * cStrideE;
  int m0 = blockIdx.x * 16;
  int n0 = blockIdx.y * 16;
  v8f acc = {};
  for (int k0 = 0; k0 < K; k0 += 32) {
    v16bf af = load_afrag(Ae + (size_t)m0 * lda + k0, lda, lane);
    v16bf bf = load_bfrag(Be + (size_t)n0 * ldb + k0, ldb, lane);
    acc = wmma_bf16(af, bf, acc);
  }
  int col = lane & 15;
  int rb  = (lane >> 4) * 8;
#pragma unroll
  for (int i = 0; i < 8; ++i) {
    size_t idx = (size_t)(m0 + rb + i) * ldc + n0 + col;
    if (accumulate) Ce[idx] += acc[i];
    else            Ce[idx] = acc[i];
  }
}

// ---- K-split skinny WMMA GEMM (for deep-K, narrow-N): atomic accumulate into C ----
__global__ __launch_bounds__(32) void gemm_bf16_skinny_ksplit(const __bf16* __restrict__ A,
                                                              int lda,
                                                              const __bf16* __restrict__ B,
                                                              int ldb,
                                                              float* __restrict__ C, int ldc,
                                                              int Kchunk) {
  int lane = threadIdx.x & 31;
  int m0 = blockIdx.x * 16;
  int n0 = blockIdx.y * 16;
  int kb = blockIdx.z * Kchunk;
  v8f acc = {};
  for (int k0 = kb; k0 < kb + Kchunk; k0 += 32) {
    v16bf af = load_afrag(A + (size_t)m0 * lda + k0, lda, lane);
    v16bf bf = load_bfrag(B + (size_t)n0 * ldb + k0, ldb, lane);
    acc = wmma_bf16(af, bf, acc);
  }
  int col = lane & 15;
  int rb  = (lane >> 4) * 8;
#pragma unroll
  for (int i = 0; i < 8; ++i)
    atomicAdd(&C[(size_t)(m0 + rb + i) * ldc + n0 + col], acc[i]);
}

// ---------------- per-expert silu(x1)*x3 with rank-32 LoRA corrections -------------
// Skips whole token when combine weight is zero (top-2 sparsity).
__global__ __launch_bounds__(256) void expert_x2_kernel(const float* __restrict__ X1B,
                                                        const float* __restrict__ X3B,
                                                        const float* __restrict__ U1,
                                                        const float* __restrict__ U3,
                                                        const __bf16* __restrict__ B1,
                                                        const __bf16* __restrict__ B3,
                                                        const float* __restrict__ cw, int e,
                                                        float* __restrict__ X2C,
                                                        __bf16* __restrict__ X2E) {
  int t = blockIdx.y;
  float w = cw[t * E + e];
  if (w == 0.f) return;  // uniform across block
  __shared__ float u1[R], u3[R];
  int tid = threadIdx.x;
  if (tid < R) {
    u1[tid] = U1[t * R + tid];
    u3[tid] = U3[t * R + tid];
  }
  __syncthreads();
  int f = blockIdx.x * 256 + tid;  // F = 56*256
  const __bf16* b1 = B1 + (size_t)f * R;
  const __bf16* b3 = B3 + (size_t)f * R;
  float d1 = 0.f, d3 = 0.f;
#pragma unroll
  for (int r = 0; r < R; ++r) {
    d1 += u1[r] * (float)b1[r];
    d3 += u3[r] * (float)b3[r];
  }
  size_t idx = (size_t)t * F + f;
  float x1 = X1B[idx] + d1;
  float x3 = X3B[idx] + d3;
  float si = x1 / (1.f + __expf(-x1));
  float x2 = si * x3;
  X2C[idx] += w * x2;
  X2E[idx] = (__bf16)x2;
}

// ---------------- host side ----------------
static inline char* bump(char*& p, size_t bytes) {
  char* r = p;
  p += (bytes + 255) & ~(size_t)255;
  return r;
}

extern "C" void kernel_launch(void* const* d_in, const int* in_sizes, int n_in,
                              void* d_out, int out_size, void* d_ws, size_t ws_size,
                              hipStream_t stream) {
  const float* x      = (const float*)d_in[0];   // (S,H)
  const float* gate_w = (const float*)d_in[1];   // (E,H)
  const float* W1     = (const float*)d_in[2];   // (F,H)
  const float* W2     = (const float*)d_in[3];   // (H,F)
  const float* W3     = (const float*)d_in[4];   // (F,H)
  const float* A1     = (const float*)d_in[5];   // (E,R,H)
  const float* B1     = (const float*)d_in[6];   // (E,F,R)
  const float* A2     = (const float*)d_in[7];   // (E,R,F)
  const float* B2     = (const float*)d_in[8];   // (E,H,R)
  const float* A3     = (const float*)d_in[9];   // (E,R,H)
  const float* B3     = (const float*)d_in[10];  // (E,F,R)

  float* out_final  = (float*)d_out;                 // (S,H)
  float* out_logits = out_final + (size_t)S * H;     // (S,E)

  // ---- carve workspace ----
  char* p = (char*)d_ws;
  float*  cw    = (float*)bump(p, (size_t)S * E * 4);
  __bf16* xb    = (__bf16*)bump(p, (size_t)S * H * 2);
  __bf16* W1b   = (__bf16*)bump(p, (size_t)F * H * 2);
  __bf16* W3b   = (__bf16*)bump(p, (size_t)F * H * 2);
  __bf16* W2b   = (__bf16*)bump(p, (size_t)H * F * 2);
  __bf16* A1b   = (__bf16*)bump(p, (size_t)E * R * H * 2);
  __bf16* A3b   = (__bf16*)bump(p, (size_t)E * R * H * 2);
  __bf16* B1b   = (__bf16*)bump(p, (size_t)E * F * R * 2);
  __bf16* B3b   = (__bf16*)bump(p, (size_t)E * F * R * 2);
  __bf16* A2b   = (__bf16*)bump(p, (size_t)E * R * F * 2);
  __bf16* B2b   = (__bf16*)bump(p, (size_t)E * H * R * 2);
  float*  X1B   = (float*)bump(p, (size_t)S * F * 4);
  float*  X3B   = (float*)bump(p, (size_t)S * F * 4);
  float*  X2C   = (float*)bump(p, (size_t)S * F * 4);
  __bf16* X2Cb  = (__bf16*)bump(p, (size_t)S * F * 2);
  __bf16* X2Eb  = (__bf16*)bump(p, (size_t)S * F * 2);
  float*  U1    = (float*)bump(p, (size_t)E * S * R * 4);
  float*  U3    = (float*)bump(p, (size_t)E * S * R * 4);
  float*  V2    = (float*)bump(p, (size_t)S * R * 4);
  __bf16* V2b   = (__bf16*)bump(p, (size_t)S * R * 2);
  float*  LORA2 = (float*)bump(p, (size_t)S * H * 4);

  auto cvt = [&](const float* in, __bf16* out, size_t n) {
    int g = (int)((n + 255) / 256);
    if (g > 4096) g = 4096;
    cvt_f32_bf16_kernel<<<g, 256, 0, stream>>>(in, out, n);
  };
  auto zero = [&](float* ptr, size_t n) {
    int g = (int)((n + 255) / 256);
    if (g > 4096) g = 4096;
    zero_f32_kernel<<<g, 256, 0, stream>>>(ptr, n);
  };

  // 1) precision conversion + accumulator init
  cvt(x,  xb,  (size_t)S * H);
  cvt(W1, W1b, (size_t)F * H);
  cvt(W3, W3b, (size_t)F * H);
  cvt(W2, W2b, (size_t)H * F);
  cvt(A1, A1b, (size_t)E * R * H);
  cvt(A3, A3b, (size_t)E * R * H);
  cvt(B1, B1b, (size_t)E * F * R);
  cvt(B3, B3b, (size_t)E * F * R);
  cvt(A2, A2b, (size_t)E * R * F);
  cvt(B2, B2b, (size_t)E * H * R);
  zero(X2C,   (size_t)S * F);
  zero(LORA2, (size_t)S * H);

  // 2) router: logits + top-2 combine weights
  router_kernel<<<S, 256, 0, stream>>>(x, gate_w, out_logits, cw);

  // 3) base projections: X1B = x@W1^T, X3B = x@W3^T
  dim3 gBig(S / 64, F / 256);
  gemm_bf16_big<<<gBig, 256, 0, stream>>>(xb, H, W1b, H, nullptr, X1B, F, H);
  gemm_bf16_big<<<gBig, 256, 0, stream>>>(xb, H, W3b, H, nullptr, X3B, F, H);

  // 4) LoRA down-projections for all experts: U1[e] = x@A1[e]^T, U3[e] = x@A3[e]^T
  dim3 gU(S / 16, R / 16, E);
  gemm_bf16_skinny<<<gU, 32, 0, stream>>>(xb, 0, H, A1b, (long)R * H, H,
                                          U1, (long)S * R, R, H, 0);
  gemm_bf16_skinny<<<gU, 32, 0, stream>>>(xb, 0, H, A3b, (long)R * H, H,
                                          U3, (long)S * R, R, H, 0);

  // 5) per-expert: silu/mul + LoRA2 path
  for (int e = 0; e < E; ++e) {
    expert_x2_kernel<<<dim3(F / 256, S), 256, 0, stream>>>(
        X1B, X3B, U1 + (size_t)e * S * R, U3 + (size_t)e * S * R,
        B1b + (size_t)e * F * R, B3b + (size_t)e * F * R, cw, e, X2C, X2Eb);
    // V2 = x2_e @ A2[e]^T  (M=S, N=R, K=F) -- K-split x8 for occupancy
    zero(V2, (size_t)S * R);
    gemm_bf16_skinny_ksplit<<<dim3(S / 16, R / 16, 8), 32, 0, stream>>>(
        X2Eb, F, A2b + (size_t)e * R * F, F, V2, R, F / 8);
    // V2b = w * V2 (bf16); inactive tokens get 0
    v2_scale_cvt_kernel<<<(S * R) / 256, 256, 0, stream>>>(V2, cw, e, V2b);
    // LORA2 += V2b @ B2[e]^T  (M=S, N=H, K=R=32 -> single WMMA per tile)
    gemm_bf16_skinny<<<dim3(S / 16, H / 16, 1), 32, 0, stream>>>(
        V2b, 0, R, B2b + (size_t)e * H * R, 0, R, LORA2, 0, H, R, 1);
  }

  // 6) final = X2C @ W2^T + LORA2
  cvt(X2C, X2Cb, (size_t)S * F);
  gemm_bf16_big<<<dim3(S / 64, H / 256), 256, 0, stream>>>(X2Cb, F, W2b, F,
                                                           LORA2, out_final, H, F);
}
